// GemPretrain_37417755083393
// MI455X (gfx1250) — matrix operand order repaired and checked by reading, not compiled, not run on hardware
//
#include <hip/hip_runtime.h>

// ---------------------------------------------------------------------------
// MI455X (gfx1250, wave32) implementation.
// GEMMs run as v_wmma_f32_16x16x32_bf16 (16x16 D tile, K=32, f32 accum).
// A-fragment per-lane layout (16-bit A 16x32, ISA 7.12.2):
//   lanes 0-15 : row M=lane,   elems 0..7 -> K=0..7,  elems 8..15 -> K=16..23
//   lanes16-31 : row M=lane-16,elems 0..7 -> K=8..15, elems 8..15 -> K=24..31
// B-fragment (32x16): lane holds column N=lane&15; lanes 0-15 K=0..7,16..23,
//   lanes 16-31 K=8..15,24..31 (per ISA 7.12.4 B layout pattern). W1 is pre-packed
//   into this order so B loads are contiguous 32B per lane.
// ---------------------------------------------------------------------------

typedef __attribute__((ext_vector_type(16))) __bf16 v16bf;
typedef __attribute__((ext_vector_type(8)))  float  v8f;

union ABFrag { uint4 q[2]; v16bf v; };

__device__ __forceinline__ unsigned short f2bf(float x) {
  unsigned int u = __float_as_uint(x);
  u += 0x7FFFu + ((u >> 16) & 1u);          // round-to-nearest-even
  return (unsigned short)(u >> 16);
}

__device__ __forceinline__ float silu(float x) {
  return x / (1.0f + __expf(-x));
}

__global__ void zero_out_kernel(float* out) { out[0] = 0.0f; }

// Repack W1 [KT x 256] f32 row-major -> bf16 in WMMA B-fragment order:
// dst[ ((kt*16 + nt)*32 + lane)*16 + i ] = W1[k][n],
//   k = kt*32 + (i&7) + (lane>=16 ? 8 : 0) + (i>=8 ? 16 : 0),  n = nt*16 + (lane&15)
__global__ __launch_bounds__(256) void pack_w1_kernel(
    const float* __restrict__ W, unsigned short* __restrict__ dst, int KT) {
  int f = blockIdx.x * 256 + threadIdx.x;
  if (f >= KT * 256) return;
  int i    =  f        & 15;
  int lane = (f >> 4)  & 31;
  int nt   = (f >> 9)  & 15;
  int kt   =  f >> 13;
  int k = kt * 32 + (i & 7) + ((lane >= 16) ? 8 : 0) + ((i >= 8) ? 16 : 0);
  int n = nt * 16 + (lane & 15);
  dst[f] = f2bf(W[k * 256 + n]);
}

// Shared GEMM body: stages a 32 x KTOT gathered-edge tile in LDS as bf16 and
// accumulates hidden[32 x 256] in per-wave WMMA accumulators acc[2][2].
template<int KTOT, int R>
__device__ __forceinline__ void edge_gemm_tile(
    const float* __restrict__ fea, const int* __restrict__ idx, int NE, int e0,
    const unsigned short* __restrict__ pw1,
    unsigned short* As /* [32 * (KTOT+8)] */, v8f acc[2][2]) {
  constexpr int KP = KTOT + 8;               // +16B row pad vs LDS banks
  const int tid = threadIdx.x;

  // ---- gather + f32->bf16 convert into LDS (coalesced float4 per row) ----
  {
    const int m = tid >> 3, sub = tid & 7;   // 32 rows x 8 threads
    const int e = min(e0 + m, NE - 1);
#pragma unroll
    for (int p = 0; p < R; ++p) {
      const int src = idx[p * NE + e];
      const float* rp = fea + (size_t)src * 256;
      unsigned short* dp = As + m * KP + p * 256;
#pragma unroll
      for (int it = 0; it < 8; ++it) {
        const int c = (sub + it * 8) * 4;
        float4 f = *(const float4*)(rp + c);
        ushort4 h;
        h.x = f2bf(f.x); h.y = f2bf(f.y); h.z = f2bf(f.z); h.w = f2bf(f.w);
        *(ushort4*)(dp + c) = h;
      }
    }
  }
  __syncthreads();

  const int lane = tid & 31;
  const int lane_lo = lane & 15, half = lane >> 4;
  const int n_base = (tid >> 5) * 32;        // wave -> 32 output columns

  for (int k0 = 0; k0 < KTOT; k0 += 32) {
    const int kt = k0 >> 5;
    ABFrag bfr[2];
#pragma unroll
    for (int ni = 0; ni < 2; ++ni) {
      const int nt = (n_base >> 4) + ni;
      bfr[ni].v = *(const v16bf*)(pw1 + (size_t)(((kt * 16 + nt) * 32 + lane) * 16));
    }
#pragma unroll
    for (int mi = 0; mi < 2; ++mi) {
      const int base = (mi * 16 + lane_lo) * KP + k0 + half * 8;
      ABFrag a;
      a.q[0] = *(const uint4*)(As + base);        // K = k0+klo .. +7
      a.q[1] = *(const uint4*)(As + base + 16);   // K = k0+16+klo .. +7
#pragma unroll
      for (int ni = 0; ni < 2; ++ni)
        acc[mi][ni] = __builtin_amdgcn_wmma_f32_16x16x32_bf16(
            false, a.v, false, bfr[ni].v, (short)0, acc[mi][ni], false, false);
    }
  }
}

// bond / angle: hidden -> SiLU -> dot(W2[256]) -> scalar, MSE accumulation
template<int KTOT, int R>
__global__ __launch_bounds__(256) void edge_mlp_scalar_kernel(
    const float* __restrict__ fea, const int* __restrict__ idx,
    const float* __restrict__ tgt, const unsigned short* __restrict__ pw1,
    const float* __restrict__ b1, const float* __restrict__ w2,
    const float* __restrict__ b2, float* __restrict__ out,
    float inv_n, int NE) {
  constexpr int KP = KTOT + 8;
  __shared__ __align__(16) unsigned short As[32 * KP];
  __shared__ float hp[32];

  const int tid = threadIdx.x;
  const int e0 = blockIdx.x * 32;
  if (tid < 32) hp[tid] = 0.0f;

  v8f acc[2][2] = {};
  edge_gemm_tile<KTOT, R>(fea, idx, NE, e0, pw1, As, acc);

  const int lane = tid & 31;
  const int lane_lo = lane & 15, half = lane >> 4;
  const int n_base = (tid >> 5) * 32;

  float b1v[2], w2v[2];
#pragma unroll
  for (int ni = 0; ni < 2; ++ni) {
    const int n = n_base + ni * 16 + lane_lo;
    b1v[ni] = b1[n];
    w2v[ni] = w2[n];
  }
#pragma unroll
  for (int mi = 0; mi < 2; ++mi) {
    float p[8];
#pragma unroll
    for (int v = 0; v < 8; ++v) {
      float x0 = acc[mi][0][v] + b1v[0];
      float x1 = acc[mi][1][v] + b1v[1];
      p[v] = silu(x0) * w2v[0] + silu(x1) * w2v[1];
    }
#pragma unroll
    for (int off = 1; off < 16; off <<= 1)
#pragma unroll
      for (int v = 0; v < 8; ++v) p[v] += __shfl_xor(p[v], off, 32);
    if (lane_lo == 0) {
      const int mb = mi * 16 + half * 8;    // rows this half-wave owns
#pragma unroll
      for (int v = 0; v < 8; ++v) atomicAdd(&hp[mb + v], p[v]);
    }
  }
  __syncthreads();
  if (tid < 32 && (e0 + tid) < NE) {
    const float pred = hp[tid] + b2[0];
    const float d = pred - tgt[e0 + tid];
    atomicAdd(out, d * d * inv_n);
  }
}

// dist: hidden -> SiLU -> [256x30] logits -> log-softmax NLL accumulation
__global__ __launch_bounds__(256) void dist_mlp_ce_kernel(
    const float* __restrict__ fea, const int* __restrict__ idx,
    const int* __restrict__ tgt, const unsigned short* __restrict__ pw1,
    const float* __restrict__ b1, const float* __restrict__ w2,
    const float* __restrict__ b2, float* __restrict__ out,
    float inv_n, int NE) {
  constexpr int KTOT = 512, KP = KTOT + 8;
  __shared__ __align__(16) unsigned short As[32 * KP];  // reused as hid f32[32][256]
  __shared__ float logits_s[32 * 30];

  const int tid = threadIdx.x;
  const int e0 = blockIdx.x * 32;

  v8f acc[2][2] = {};
  edge_gemm_tile<KTOT, 2>(fea, idx, NE, e0, pw1, As, acc);

  const int lane = tid & 31;
  const int lane_lo = lane & 15, half = lane >> 4;
  const int n_base = (tid >> 5) * 32;

  float b1v[2];
#pragma unroll
  for (int ni = 0; ni < 2; ++ni) b1v[ni] = b1[n_base + ni * 16 + lane_lo];

  __syncthreads();                     // all A-tile reads done; reuse LDS
  float* hid = (float*)As;
#pragma unroll
  for (int mi = 0; mi < 2; ++mi)
#pragma unroll
    for (int ni = 0; ni < 2; ++ni) {
      const int n = n_base + ni * 16 + lane_lo;
#pragma unroll
      for (int v = 0; v < 8; ++v) {
        const int m = mi * 16 + half * 8 + v;
        hid[m * 256 + n] = silu(acc[mi][ni][v] + b1v[ni]);
      }
    }
  __syncthreads();

  for (int t = tid; t < 32 * 30; t += 256) {
    const int row = t / 30, cls = t - row * 30;
    float s = b2[cls];
    for (int c = 0; c < 256; ++c) s += hid[row * 256 + c] * w2[c * 30 + cls];
    logits_s[t] = s;
  }
  __syncthreads();

  if (tid < 32 && (e0 + tid) < NE) {
    const int row = tid;
    float mx = -1e30f;
    for (int c = 0; c < 30; ++c) mx = fmaxf(mx, logits_s[row * 30 + c]);
    float se = 0.0f;
    for (int c = 0; c < 30; ++c) se += __expf(logits_s[row * 30 + c] - mx);
    const float lse = mx + __logf(se);
    const int tc = tgt[e0 + row];
    const float nll = lse - logits_s[row * 30 + tc];
    atomicAdd(out, nll * inv_n);
  }
}

extern "C" void kernel_launch(void* const* d_in, const int* in_sizes, int n_in,
                              void* d_out, int out_size, void* d_ws, size_t ws_size,
                              hipStream_t stream) {
  const float* fea  = (const float*)d_in[0];
  const int*   bidx = (const int*)d_in[1];
  const float* btgt = (const float*)d_in[2];
  const int*   aidx = (const int*)d_in[3];
  const float* atgt = (const float*)d_in[4];
  const int*   didx = (const int*)d_in[5];
  const int*   dtgt = (const int*)d_in[6];
  const float* bW1 = (const float*)d_in[7];
  const float* bb1 = (const float*)d_in[8];
  const float* bW2 = (const float*)d_in[9];
  const float* bb2 = (const float*)d_in[10];
  const float* aW1 = (const float*)d_in[11];
  const float* ab1 = (const float*)d_in[12];
  const float* aW2 = (const float*)d_in[13];
  const float* ab2 = (const float*)d_in[14];
  const float* dW1 = (const float*)d_in[15];
  const float* db1 = (const float*)d_in[16];
  const float* dW2 = (const float*)d_in[17];
  const float* db2 = (const float*)d_in[18];

  const int NB = in_sizes[2];   // 200000
  const int NA = in_sizes[4];   // 200000
  const int ND = in_sizes[6];   // 300000

  // Workspace: packed bf16 W1s (bond 256KB, angle 384KB, dist 256KB)
  unsigned short* pb = (unsigned short*)d_ws;
  unsigned short* pa = pb + 512 * 256;
  unsigned short* pd = pa + 768 * 256;

  float* out = (float*)d_out;
  zero_out_kernel<<<1, 1, 0, stream>>>(out);

  pack_w1_kernel<<<(512 * 256 + 255) / 256, 256, 0, stream>>>(bW1, pb, 512);
  pack_w1_kernel<<<(768 * 256 + 255) / 256, 256, 0, stream>>>(aW1, pa, 768);
  pack_w1_kernel<<<(512 * 256 + 255) / 256, 256, 0, stream>>>(dW1, pd, 512);

  edge_mlp_scalar_kernel<512, 2><<<(NB + 31) / 32, 256, 0, stream>>>(
      fea, bidx, btgt, pb, bb1, bW2, bb2, out, 1.0f / (float)NB, NB);
  edge_mlp_scalar_kernel<768, 3><<<(NA + 31) / 32, 256, 0, stream>>>(
      fea, aidx, atgt, pa, ab1, aW2, ab2, out, 1.0f / (float)NA, NA);
  dist_mlp_ce_kernel<<<(ND + 31) / 32, 256, 0, stream>>>(
      fea, didx, dtgt, pd, db1, dW2, db2, out, 1.0f / (float)ND, ND);
}
